// Static_Sink_17540646437048
// MI455X (gfx1250) — matrix operand order repaired and checked by reading, not compiled, not run on hardware
//
#include <hip/hip_runtime.h>

#define NNODES 50000
#define DIMX   1024
#define NHEAD  16
#define NKV    128
#define HDIM   64

#define MT 80      // GEMM M-tile per block (50000 = 625*80)
#define KC 128     // GEMM K-chunk staged in LDS per TDM transfer

typedef __attribute__((ext_vector_type(16))) __bf16 v16bf;
typedef __attribute__((ext_vector_type(8)))  float  v8f;
typedef __attribute__((ext_vector_type(4)))  unsigned int u32x4;
typedef __attribute__((ext_vector_type(8)))  int i32x8;
typedef __attribute__((ext_vector_type(4)))  int i32x4;

struct U8x32 { uint4 lo, hi; };

static __device__ __forceinline__ v16bf make_frag(uint4 lo, uint4 hi) {
  U8x32 u{lo, hi};
  return __builtin_bit_cast(v16bf, u);
}

// fp32 -> bf16, round-to-nearest-even
static __device__ __forceinline__ unsigned short f2bf(float f) {
  unsigned int u = __float_as_uint(f);
  u += 0x7FFFu + ((u >> 16) & 1u);
  return (unsigned short)(u >> 16);
}

static __device__ __forceinline__ float sigmoidf_(float x) {
  return 1.0f / (1.0f + __expf(-x));
}

template <int N>
static __device__ __forceinline__ void wait_tensor() {
#if __has_builtin(__builtin_amdgcn_s_wait_tensorcnt)
  __builtin_amdgcn_s_wait_tensorcnt((short)N);
#else
  if (N == 0) asm volatile("s_wait_tensorcnt 0x0" ::: "memory");
  else        asm volatile("s_wait_tensorcnt 0x1" ::: "memory");
#endif
}

// Issue a TDM 2-D tile load: `rows` rows of `row_elems` 2-byte elements,
// source row stride `row_stride_elems`, into LDS at byte offset lds_off
// (rows stored contiguously, row-major). Uniform (SGPR) descriptor.
static __device__ __forceinline__ void tdm_load_tile(const unsigned short* gsrc,
                                                     unsigned int lds_off,
                                                     unsigned int rows,
                                                     unsigned int row_elems,
                                                     unsigned int row_stride_elems,
                                                     unsigned int tensor_rows) {
  unsigned long long ga = (unsigned long long)gsrc;
  u32x4 g0;
  g0.x = 0x1u;                                    // count=1 (valid), user mode
  g0.y = lds_off;                                 // lds_addr (bytes)
  g0.z = (unsigned int)ga;                        // global_addr[31:0]
  g0.w = (unsigned int)((ga >> 32) & 0x1FFFFFFull) | 0x80000000u; // addr[56:32]|type=2
  i32x8 g1;
  g1[0] = (int)(1u << 16);                        // wg_mask=0, data_size=1 (2B)
  g1[1] = (int)(row_stride_elems << 16);          // tensor_dim0[15:0] in [31:16]
  g1[2] = (int)((row_stride_elems >> 16) | (tensor_rows << 16)); // dim0 hi | dim1 lo
  g1[3] = (int)(((tensor_rows >> 16) & 0xFFFFu) | (row_elems << 16)); // dim1 hi | tile_dim0
  g1[4] = (int)(rows & 0xFFFFu);                  // tile_dim1 (tile_dim2=0)
  g1[5] = (int)row_stride_elems;                  // tensor_dim0_stride[31:0]
  g1[6] = 0;                                      // stride hi, dim1_stride lo
  g1[7] = 0;
  i32x4 gz = {0, 0, 0, 0};
  asm volatile("tensor_load_to_lds %0, %1, %2, %3"
               :: "s"(g0), "s"(g1), "s"(gz), "s"(gz)
               : "memory");
}

// ---------------------------------------------------------------------------
// Kernel 0: fp32 -> bf16 conversion (grid-stride)
// ---------------------------------------------------------------------------
__global__ __launch_bounds__(256) void k_conv_bf16(const float* __restrict__ src,
                                                   unsigned short* __restrict__ dst,
                                                   long n) {
  long i = (long)blockIdx.x * blockDim.x + threadIdx.x;
  long stride = (long)gridDim.x * blockDim.x;
  for (; i < n; i += stride) dst[i] = f2bf(src[i]);
}

// ---------------------------------------------------------------------------
// Kernel 1: C = Xbf @ Wbf^T + b   (X: N x 1024, W: 1024 x 1024 row-major)
// Block = 8 waves; block tile = 80 (M) x 128 (N); each wave: 5 M-subtiles,
// so one B fragment feeds 5 WMMAs. A tiles (80x128 bf16 = 20KB) are staged
// into LDS by the Tensor Data Mover, double-buffered on TENSORcnt.
// A-fragment LDS loads are hoisted ahead of the WMMA burst so they clause
// behind a single s_wait_dscnt.
// ---------------------------------------------------------------------------
__global__ __launch_bounds__(256) void k_gemm_xw(const unsigned short* __restrict__ xb,
                                                 const unsigned short* __restrict__ wb,
                                                 const float* __restrict__ bias,
                                                 unsigned short* __restrict__ out_bf,
                                                 float* __restrict__ out_f32) {
  __shared__ __align__(16) unsigned short As[2][MT * KC];  // 2 x 20 KB
  const int m0   = blockIdx.x * MT;
  const int n0b  = blockIdx.y * 128;
  const int t    = threadIdx.x;
  const int wave = t >> 5;
  const int lane = t & 31;
  const int col  = lane & 15;
  const int hi   = lane >> 4;
  const int c    = n0b + wave * 16 + col;   // output column for this lane
  const int H8   = hi * 8;

  const unsigned int lds0 =
      (unsigned int)(unsigned long long)(__attribute__((address_space(3))) unsigned short*)&As[0][0];
  const unsigned int lds1 =
      (unsigned int)(unsigned long long)(__attribute__((address_space(3))) unsigned short*)&As[1][0];

  v8f zero = {};
  v8f acc[5] = {zero, zero, zero, zero, zero};

  // prefetch chunk 0 into buffer 0
  if (wave == 0) {
    tdm_load_tile(&xb[(long)m0 * DIMX], lds0, MT, KC, DIMX, (unsigned)NNODES);
  }

#pragma unroll 1
  for (int ch = 0; ch < DIMX / KC; ++ch) {
    const int buf = ch & 1;
    if (wave == 0) {
      if (ch < DIMX / KC - 1) {
        tdm_load_tile(&xb[(long)m0 * DIMX + (ch + 1) * KC],
                      buf ? lds0 : lds1, MT, KC, DIMX, (unsigned)NNODES);
        wait_tensor<1>();   // chunk `ch` complete (TDM ops in-order per wave)
      } else {
        wait_tensor<0>();
      }
    }
    __syncthreads();

    const unsigned short* wbase = &wb[(long)c * DIMX + ch * KC];
#pragma unroll
    for (int ks = 0; ks < KC / 32; ++ks) {
      const unsigned short* bp = wbase + ks * 32 + H8;
      v16bf b = make_frag(*(const uint4*)bp, *(const uint4*)(bp + 16));
      v16bf a[5];
#pragma unroll
      for (int mt = 0; mt < 5; ++mt) {
        const unsigned short* ap = &As[buf][(mt * 16 + col) * KC + ks * 32 + H8];
        a[mt] = make_frag(*(const uint4*)ap, *(const uint4*)(ap + 16));
      }
#pragma unroll
      for (int mt = 0; mt < 5; ++mt) {
        acc[mt] = __builtin_amdgcn_wmma_f32_16x16x32_bf16(false, a[mt], false, b,
                                                          (short)0, acc[mt],
                                                          false, false);
      }
    }
    __syncthreads();   // all waves done reading buf before it is refilled
  }

  const float bcol = bias[c];
#pragma unroll
  for (int mt = 0; mt < 5; ++mt) {
#pragma unroll
    for (int j = 0; j < 8; ++j) {
      int M = mt * 16 + j + hi * 8;
      float val = acc[mt][j] + bcol;
      long off = (long)(m0 + M) * DIMX + c;
      if (out_f32) out_f32[off] = val;
      if (out_bf)  out_bf[off]  = f2bf(val);
    }
  }
}

// ---------------------------------------------------------------------------
// Kernel 2: attn[n,h,r] = (q[n,h,:] . key[r,h,:]) / 8 + bias(node_index)
// Block handles 16 nodes x 128 slots for one head (8 waves x 16 slots).
// Also writes per-(n,h) softmax stats (rowmax, rowsum).
// ---------------------------------------------------------------------------
__global__ __launch_bounds__(256) void k_attn(const unsigned short* __restrict__ qb,
                                              const unsigned short* __restrict__ keyb,
                                              const float* __restrict__ nidx,
                                              const float* __restrict__ Wb,
                                              const float* __restrict__ bb,
                                              float* __restrict__ attn,
                                              float* __restrict__ smax,
                                              float* __restrict__ ssum) {
  __shared__ float Att[16 * NKV];            // 8 KB
  const int m0   = blockIdx.x * 16;
  const int h    = blockIdx.y;
  const int t    = threadIdx.x;
  const int wave = t >> 5;
  const int lane = t & 31;
  const int col  = lane & 15;
  const int hi   = lane >> 4;
  const int r    = wave * 16 + col;
  const int H8   = hi * 8;

  // load both K-step fragments up front, then issue both WMMAs
  v16bf a[2], b[2];
#pragma unroll
  for (int kk = 0; kk < 2; ++kk) {
    const unsigned short* ap = &qb[(long)(m0 + col) * DIMX + h * HDIM + kk * 32 + H8];
    a[kk] = make_frag(*(const uint4*)ap, *(const uint4*)(ap + 16));
    const unsigned short* bp = &keyb[((long)r * NHEAD + h) * HDIM + kk * 32 + H8];
    b[kk] = make_frag(*(const uint4*)bp, *(const uint4*)(bp + 16));
  }
  v8f acc = {};
#pragma unroll
  for (int kk = 0; kk < 2; ++kk) {
    acc = __builtin_amdgcn_wmma_f32_16x16x32_bf16(false, a[kk], false, b[kk],
                                                  (short)0, acc, false, false);
  }

  const float* wrow = &Wb[((long)h * NKV + r) * 4];
  float w0 = wrow[0], w1 = wrow[1], w2 = wrow[2], w3 = wrow[3];
  float bcol = bb[h * NKV + r];
#pragma unroll
  for (int j = 0; j < 8; ++j) {
    int M = j + hi * 8;
    const float* ni = &nidx[(long)(m0 + M) * 4];
    float bias = bcol + w0 * ni[0] + w1 * ni[1] + w2 * ni[2] + w3 * ni[3];
    Att[M * NKV + r] = acc[j] * 0.125f + bias;  // 1/sqrt(64)
  }
  __syncthreads();

#pragma unroll
  for (int i = 0; i < 8; ++i) {
    int flat = t + i * 256;
    int rw = flat >> 7, cc = flat & 127;
    attn[((long)(m0 + rw) * NHEAD + h) * NKV + cc] = Att[rw * NKV + cc];
  }

  if (t < 16) {
    float m = -3.4e38f;
    for (int i = 0; i < NKV; ++i) m = fmaxf(m, Att[t * NKV + i]);
    float s = 0.0f;
    for (int i = 0; i < NKV; ++i) s += __expf(Att[t * NKV + i] - m);
    smax[(long)(m0 + t) * NHEAD + h] = m;
    ssum[(long)(m0 + t) * NHEAD + h] = s;
  }
}

// ---------------------------------------------------------------------------
// Kernel 3: v[r,h,d] += sum_n sigmoid(attn[n,h,r]) * xv[n,h,d]
// ---------------------------------------------------------------------------
__global__ __launch_bounds__(256) void k_vacc(const float* __restrict__ attn,
                                              const float* __restrict__ xv,
                                              float* __restrict__ vbuf) {
  const int h  = blockIdx.y;
  const int n0 = blockIdx.x * 512;
  int nend = n0 + 512; if (nend > NNODES) nend = NNODES;
  const int t  = threadIdx.x;
  const int r  = t & 127;
  const int d0 = (t >> 7) * 32;

  float acc[32];
#pragma unroll
  for (int i = 0; i < 32; ++i) acc[i] = 0.0f;

  for (int n = n0; n < nend; ++n) {
    float a = attn[((long)n * NHEAD + h) * NKV + r];
    float s = sigmoidf_(a);
    const float* xp = &xv[(long)n * DIMX + h * HDIM + d0];
#pragma unroll
    for (int i = 0; i < 32; i += 4) {
      float4 x4 = *(const float4*)&xp[i];
      acc[i + 0] += s * x4.x;
      acc[i + 1] += s * x4.y;
      acc[i + 2] += s * x4.z;
      acc[i + 3] += s * x4.w;
    }
  }
  float* vp = &vbuf[((long)r * NHEAD + h) * HDIM + d0];
#pragma unroll
  for (int i = 0; i < 32; ++i) atomicAdd(&vp[i], acc[i]);
}

// ---------------------------------------------------------------------------
// Kernel 4: sinkhorn pass (factored form). E = exp(softmax(attn)/eps).
// phase 0: G += sum_n E ; phase 1,2: u=1/(N*sum_r E*w), G += E*u ;
// phase 3: u_out = 1/(N*sum_r E*w)
// ---------------------------------------------------------------------------
__global__ __launch_bounds__(256) void k_sink(const float* __restrict__ attn,
                                              const float* __restrict__ smax,
                                              const float* __restrict__ ssum,
                                              const float* __restrict__ w,
                                              float* __restrict__ G,
                                              float* __restrict__ u_out,
                                              int phase) {
  __shared__ float Gl[NHEAD * NKV];  // 8 KB
  const int t = threadIdx.x;
  const long id = (long)blockIdx.x * 256 + t;  // n*NHEAD + h
  const int h = (int)(id & (NHEAD - 1));

  if (phase != 3) {
#pragma unroll
    for (int i = 0; i < 8; ++i) Gl[t + i * 256] = 0.0f;
    __syncthreads();
  }

  const float* ap = &attn[id * NKV];
  const float mx = smax[id];
  const float inv = 1.0f / ssum[id];

  float uval = 1.0f;
  if (phase >= 1) {
    float s = 0.0f;
    for (int r = 0; r < NKV; ++r) {
      float E = __expf(__expf(ap[r] - mx) * inv * 20.0f);  // 1/eps = 20
      s += E * w[h * NKV + r];
    }
    uval = 1.0f / ((float)NNODES * s);
  }
  if (phase == 3) { u_out[id] = uval; return; }

  for (int r = 0; r < NKV; ++r) {
    float E = __expf(__expf(ap[r] - mx) * inv * 20.0f);
    atomicAdd(&Gl[h * NKV + r], E * uval);
  }
  __syncthreads();
#pragma unroll
  for (int i = 0; i < 8; ++i) {
    int idx = t + i * 256;
    atomicAdd(&G[idx], Gl[idx]);
  }
}

__global__ __launch_bounds__(256) void k_finw(float* __restrict__ G,
                                              float* __restrict__ w) {
  int i = blockIdx.x * 256 + threadIdx.x;
  if (i < NHEAD * NKV) {
    w[i] = 1.0f / ((float)NKV * G[i]);
    G[i] = 0.0f;
  }
}

// ---------------------------------------------------------------------------
// Kernel 5: out = sigmoid(alpha)*xv + sigmoid(beta) * N*u * sum_r (E*w) v[r]
// ---------------------------------------------------------------------------
__global__ __launch_bounds__(256) void k_out(const float* __restrict__ attn,
                                             const float* __restrict__ smax,
                                             const float* __restrict__ ssum,
                                             const float* __restrict__ u,
                                             const float* __restrict__ w,
                                             const float* __restrict__ vbuf,
                                             const float* __restrict__ xv,
                                             const float* __restrict__ alpha,
                                             const float* __restrict__ beta,
                                             float* __restrict__ out) {
  const long id = (long)blockIdx.x * 256 + threadIdx.x;  // n*NHEAD + h
  const int h = (int)(id & (NHEAD - 1));
  const long n = id >> 4;

  const float* ap = &attn[id * NKV];
  const float mx = smax[id];
  const float inv = 1.0f / ssum[id];

  float acc[HDIM];
#pragma unroll
  for (int d = 0; d < HDIM; ++d) acc[d] = 0.0f;

  for (int r = 0; r < NKV; ++r) {
    float E = __expf(__expf(ap[r] - mx) * inv * 20.0f);
    float cc = E * w[h * NKV + r];
    const float* vp = &vbuf[((long)r * NHEAD + h) * HDIM];
#pragma unroll
    for (int d = 0; d < HDIM; d += 4) {
      float4 v4 = *(const float4*)&vp[d];
      acc[d + 0] += cc * v4.x;
      acc[d + 1] += cc * v4.y;
      acc[d + 2] += cc * v4.z;
      acc[d + 3] += cc * v4.w;
    }
  }

  const float sa = sigmoidf_(alpha[h]);
  const float sb = sigmoidf_(beta[h]);
  const float scale = sb * (float)NNODES * u[id];
  const float* xp = &xv[n * DIMX + h * HDIM];
  float* op = &out[n * DIMX + h * HDIM];
#pragma unroll
  for (int d = 0; d < HDIM; ++d) op[d] = sa * xp[d] + scale * acc[d];
}

// ---------------------------------------------------------------------------
extern "C" void kernel_launch(void* const* d_in, const int* in_sizes, int n_in,
                              void* d_out, int out_size, void* d_ws, size_t ws_size,
                              hipStream_t stream) {
  (void)in_sizes; (void)n_in; (void)out_size; (void)ws_size;
  const float* x     = (const float*)d_in[0];
  const float* nidx  = (const float*)d_in[1];
  const float* Wq    = (const float*)d_in[2];
  const float* bq    = (const float*)d_in[3];
  const float* key_p = (const float*)d_in[4];
  const float* Wv    = (const float*)d_in[5];
  const float* bv    = (const float*)d_in[6];
  const float* alpha = (const float*)d_in[7];
  const float* beta  = (const float*)d_in[8];
  const float* Wb    = (const float*)d_in[9];
  const float* bb    = (const float*)d_in[10];
  float* out = (float*)d_out;

  char* ws = (char*)d_ws;
  size_t off = 0;
  auto alloc = [&](size_t bytes) -> void* {
    void* p = (void*)(ws + off);
    off += (bytes + 255) & ~(size_t)255;
    return p;
  };
  unsigned short* x_bf   = (unsigned short*)alloc((size_t)NNODES * DIMX * 2);
  unsigned short* q_bf   = (unsigned short*)alloc((size_t)NNODES * DIMX * 2);
  unsigned short* wq_bf  = (unsigned short*)alloc((size_t)DIMX * DIMX * 2);
  unsigned short* wv_bf  = (unsigned short*)alloc((size_t)DIMX * DIMX * 2);
  unsigned short* key_bf = (unsigned short*)alloc((size_t)NKV * NHEAD * HDIM * 2);
  float* xv   = (float*)alloc((size_t)NNODES * DIMX * 4);
  float* attn = (float*)alloc((size_t)NNODES * NHEAD * NKV * 4);
  float* smax = (float*)alloc((size_t)NNODES * NHEAD * 4);
  float* ssum = (float*)alloc((size_t)NNODES * NHEAD * 4);
  float* ubuf = (float*)alloc((size_t)NNODES * NHEAD * 4);
  float* vbuf = (float*)alloc((size_t)NKV * NHEAD * HDIM * 4);
  float* Gbuf = (float*)alloc((size_t)NHEAD * NKV * 4);
  float* wbuf = (float*)alloc((size_t)NHEAD * NKV * 4);

  (void)hipMemsetAsync(Gbuf, 0, (size_t)NHEAD * NKV * 4, stream);
  (void)hipMemsetAsync(vbuf, 0, (size_t)NKV * NHEAD * HDIM * 4, stream);

  k_conv_bf16<<<2048, 256, 0, stream>>>(x, x_bf, (long)NNODES * DIMX);
  k_conv_bf16<<<1024, 256, 0, stream>>>(Wq, wq_bf, (long)DIMX * DIMX);
  k_conv_bf16<<<1024, 256, 0, stream>>>(Wv, wv_bf, (long)DIMX * DIMX);
  k_conv_bf16<<<64, 256, 0, stream>>>(key_p, key_bf, (long)NKV * NHEAD * HDIM);

  dim3 gg(NNODES / MT, DIMX / 128);   // 625 x 8
  k_gemm_xw<<<gg, 256, 0, stream>>>(x_bf, wq_bf, bq, q_bf, (float*)nullptr);
  k_gemm_xw<<<gg, 256, 0, stream>>>(x_bf, wv_bf, bv, (unsigned short*)nullptr, xv);

  dim3 ga(NNODES / 16, NHEAD);
  k_attn<<<ga, 256, 0, stream>>>(q_bf, key_bf, nidx, Wb, bb, attn, smax, ssum);

  dim3 gv((NNODES + 511) / 512, NHEAD);
  k_vacc<<<gv, 256, 0, stream>>>(attn, xv, vbuf);

  int nblk = (NNODES * NHEAD) / 256;  // 3125
  k_sink<<<nblk, 256, 0, stream>>>(attn, smax, ssum, wbuf, Gbuf, ubuf, 0);
  k_finw<<<8, 256, 0, stream>>>(Gbuf, wbuf);
  k_sink<<<nblk, 256, 0, stream>>>(attn, smax, ssum, wbuf, Gbuf, ubuf, 1);
  k_finw<<<8, 256, 0, stream>>>(Gbuf, wbuf);
  k_sink<<<nblk, 256, 0, stream>>>(attn, smax, ssum, wbuf, Gbuf, ubuf, 2);
  k_finw<<<8, 256, 0, stream>>>(Gbuf, wbuf);
  k_sink<<<nblk, 256, 0, stream>>>(attn, smax, ssum, wbuf, Gbuf, ubuf, 3);

  k_out<<<nblk, 256, 0, stream>>>(attn, smax, ssum, ubuf, wbuf, vbuf, xv,
                                  alpha, beta, out);
}
